// ATTfold_23124103922187
// MI455X (gfx1250) — matrix operand order repaired and checked by reading, not compiled, not run on hardware
//
#include <hip/hip_runtime.h>
#include <hip/hip_bf16.h>

typedef __attribute__((ext_vector_type(16))) _Float16 v16h;
typedef __attribute__((ext_vector_type(8)))  float    v8f;
typedef __attribute__((ext_vector_type(4)))  unsigned int u32x4;
typedef __attribute__((ext_vector_type(8)))  int          i32x8;
typedef __attribute__((ext_vector_type(4)))  int          i32x4;

#define BB 2
#define LL 512
#define DD 32
#define EE 64
#define HH 2
#define DHD 32
#define FFD 2048
#define NSTEPS 20
#define LSQ (LL*LL)           // 262144
#define BLL (BB*LL*LL)        // 524288

#if defined(__has_builtin)
#if __has_builtin(__builtin_amdgcn_tensor_load_to_lds) && __has_builtin(__builtin_amdgcn_s_wait_tensorcnt)
#define HAVE_TDM 1
#endif
#endif

// ---------------------------------------------------------------------------
// Generic WMMA f16 GEMM: C[M,N] = A[M,K] * B[K,N] (or A*B^T), fp32 accum.
// Batched: per z, offX = (z/h2)*sX1 + (z%h2)*sX2.
// Block = 256 threads = 8 waves; block tile 64x32; wave tile 16x16; K step 32.
// A tile staged via Tensor Data Mover (tensor_load_to_lds) when available.
// B tile staged TRANSPOSED in LDS (Bs[n][k]) so WMMA B-fragments are two
// contiguous ds_load_b128 per lane instead of 16 strided ds_load_u16.
// ---------------------------------------------------------------------------
template<bool TRANSB>
__global__ __launch_bounds__(256)
void k_gemm(const _Float16* __restrict__ A, int lda, long sA1, long sA2,
            const _Float16* __restrict__ Bm, int ldb, long sB1, long sB2,
            float* Cf, _Float16* Ch, int ldc, long sC1, long sC2,
            const float* bias, int h2, int M, int N, int K, int relu) {
  __shared__ _Float16 As[64 * 32];   // row-major [m][k]
  __shared__ _Float16 Bs[32 * 32];   // TRANSPOSED: [n][k]
  int z = blockIdx.z;
  const _Float16* Ab = A  + (long)(z / h2) * sA1 + (long)(z % h2) * sA2;
  const _Float16* Bb = Bm + (long)(z / h2) * sB1 + (long)(z % h2) * sB2;
  long coff = (long)(z / h2) * sC1 + (long)(z % h2) * sC2;
  int m0 = blockIdx.x * 64, n0 = blockIdx.y * 32;
  int tid = threadIdx.x, lane = tid & 31, wid = tid >> 5;
  int wm = wid & 3, wn = wid >> 2;
  v8f acc = {};
#ifdef HAVE_TDM
  unsigned ldsA = (unsigned)(unsigned long long)(&As[0]);           // LDS aperture: addr[31:0]
  unsigned long long gaRow = (unsigned long long)(const void*)(Ab + (long)m0 * lda);
#endif
  for (int k0 = 0; k0 < K; k0 += 32) {
#ifdef HAVE_TDM
    // ---- A tile (64 rows x 32 halves) via Tensor Data Mover, issued by wave 0 ----
    if (wid == 0) {
      unsigned long long ga = gaRow + (unsigned long long)k0 * 2;   // byte address of tile start
      u32x4 g0;
      g0.x = 1u;                                                    // count=1, user mode
      g0.y = ldsA;                                                  // lds_addr (bytes)
      g0.z = (unsigned)ga;                                          // global_addr[31:0]
      g0.w = (unsigned)((ga >> 32) & 0x01FFFFFFull) | (2u << 30);   // global_addr[56:32] | type=2
      i32x8 g1;
      unsigned td0 = (unsigned)K, td1 = (unsigned)M;
      g1[0] = (int)(1u << 16);                                      // data_size=1 -> 2 bytes
      g1[1] = (int)((td0 & 0xFFFFu) << 16);                         // tensor_dim0[15:0]
      g1[2] = (int)((td0 >> 16) | ((td1 & 0xFFFFu) << 16));         // tensor_dim0[31:16]|tensor_dim1[15:0]
      g1[3] = (int)((td1 >> 16) | (32u << 16));                     // tensor_dim1[31:16]|tile_dim0=32
      g1[4] = 64;                                                   // tile_dim1=64, tile_dim2=0
      g1[5] = lda;                                                  // tensor_dim0_stride[31:0]
      g1[6] = 0;
      g1[7] = 0;
      i32x4 zz = {0, 0, 0, 0};
#if __clang_major__ >= 23
      i32x8 zz8 = {0, 0, 0, 0, 0, 0, 0, 0};
      __builtin_amdgcn_tensor_load_to_lds(g0, g1, zz, zz, zz8, 0);
#else
      __builtin_amdgcn_tensor_load_to_lds(g0, g1, zz, zz, 0);
#endif
      __builtin_amdgcn_s_wait_tensorcnt(0);
    }
#else
    // ---- manual A staging: 256 x 16B vector copies ----
    {
      int r = tid >> 2, c4 = (tid & 3) * 8;
      u32x4 val = *(const u32x4*)&Ab[(long)(m0 + r) * lda + k0 + c4];
      *(u32x4*)&As[r * 32 + c4] = val;
    }
#endif
    // ---- B staging into transposed LDS tile ----
    if (TRANSB) {
      if (tid < 128) {                       // 128 x 8-half vector copies
        int c = tid >> 2, q = (tid & 3) * 8;
        u32x4 val = *(const u32x4*)&Bb[(long)(n0 + c) * ldb + k0 + q];
        *(u32x4*)&Bs[c * 32 + q] = val;
      }
    } else {
#pragma unroll
      for (int i = 0; i < 4; ++i) {
        int idx = tid * 4 + i;
        int k = idx >> 5, c = idx & 31;
        Bs[c * 32 + k] = Bb[(long)(k0 + k) * ldb + n0 + c];
      }
    }
    if (k0 + 32 < K)
      __builtin_prefetch(&Ab[(long)(m0 + (tid & 63)) * lda + k0 + 32], 0, 1);
    __syncthreads();
    // ---- fragments: A lane<16 -> row=lane, K {0..7,16..23}; lane>=16 -> K {8..15,24..31}
    union { v16h h; u32x4 q[2]; } afu, bfu;
    {
      int kb = (lane >> 4) * 8;
      const u32x4* ap = (const u32x4*)&As[(wm * 16 + (lane & 15)) * 32 + kb];
      afu.q[0] = ap[0];                      // halves kb .. kb+7
      afu.q[1] = ap[2];                      // halves kb+16 .. kb+23
      int col = wn * 16 + (lane & 15);
      int kb2 = (lane >> 4) * 8;
      const u32x4* bp = (const u32x4*)&Bs[col * 32 + kb2];
      bfu.q[0] = bp[0];                      // K kb2 .. kb2+7
      bfu.q[1] = bp[2];                      // K kb2+16 .. kb2+23
    }
    acc = __builtin_amdgcn_wmma_f32_16x16x32_f16(false, afu.h, false, bfu.h,
                                                 (short)0, acc, false, false);
    __syncthreads();
  }
  int n  = n0 + wn * 16 + (lane & 15);
  int mb = m0 + wm * 16 + ((lane >> 4) * 8);
  float bv = bias ? bias[n] : 0.f;
#pragma unroll
  for (int r = 0; r < 8; ++r) {
    float val = acc[r] + bv;
    if (relu) val = fmaxf(val, 0.f);
    long off = coff + (long)(mb + r) * ldc + n;
    if (Cf) Cf[off] = val;
    if (Ch) Ch[off] = (_Float16)val;
  }
}

// ---------------------------------------------------------------------------
__global__ void k_cvt_f16(_Float16* dst, const float* src, int n) {
  int i = blockIdx.x * 256 + threadIdx.x;
  if (i < n) dst[i] = (_Float16)src[i];
}
__global__ void k_zero(float* p, int n) {
  int i = blockIdx.x * 256 + threadIdx.x;
  if (i < n) p[i] = 0.f;
}
__global__ void k_copy(float* dst, const float* src, int n) {
  int i = blockIdx.x * 256 + threadIdx.x;
  if (i < n) dst[i] = src[i];
}

// PE-net MLP: 111 -> 160 -> 160 -> 32, one block per token (160 threads)
__global__ void k_pe_mlp(const float* pe, const float* w1, const float* b1,
                         const float* w2, const float* b2,
                         const float* w3, const float* b3, float* pos) {
  __shared__ float pin[111], h1[160], h2[160];
  int t = blockIdx.x, j = threadIdx.x;
  const float* xr = pe + (long)t * 111;
  for (int i = j; i < 111; i += 160) pin[i] = xr[i];
  __syncthreads();
  float s = b1[j];
  for (int i = 0; i < 111; ++i) s += pin[i] * w1[i * 160 + j];
  h1[j] = fmaxf(s, 0.f);
  __syncthreads();
  s = b2[j];
  for (int i = 0; i < 160; ++i) s += h1[i] * w2[i * 160 + j];
  h2[j] = fmaxf(s, 0.f);
  __syncthreads();
  if (j < 32) {
    s = b3[j];
    for (int i = 0; i < 160; ++i) s += h2[i] * w3[i * 32 + j];
    pos[(long)t * 32 + j] = s;
  }
}

// dilated conv1d (k=9, dil=2, pad=8) on one-hot seq -> (B,32,512)
__global__ void k_conv(const float* seq, const float* cw, const float* cb, float* out) {
  int idx = blockIdx.x * 256 + threadIdx.x;
  if (idx >= BB * DD * LL) return;
  int l = idx & 511, d = (idx >> 9) & 31, b = idx >> 14;
  float s = cb[d];
  for (int kk = 0; kk < 9; ++kk) {
    int p = l + 2 * kk - 8;
    if (p < 0 || p >= LL) continue;
    const float* sp = seq + ((long)b * LL + p) * 4;
    const float* wp = cw + d * 36 + kk;
    for (int c = 0; c < 4; ++c) s += sp[c] * wp[c * 9];
  }
  out[idx] = s;
}

__global__ void k_bn1d_stats(const float* xin, float* mean, float* inv) {
  int d = blockIdx.x;
  float s = 0.f, ss = 0.f;
  for (int i = threadIdx.x; i < BB * LL; i += 256) {
    int b = i >> 9, l = i & 511;
    float v = xin[((long)b * DD + d) * LL + l];
    s += v; ss += v * v;
  }
  __shared__ float r1[256], r2[256];
  r1[threadIdx.x] = s; r2[threadIdx.x] = ss; __syncthreads();
  for (int o = 128; o > 0; o >>= 1) {
    if (threadIdx.x < o) { r1[threadIdx.x] += r1[threadIdx.x + o]; r2[threadIdx.x] += r2[threadIdx.x + o]; }
    __syncthreads();
  }
  if (threadIdx.x == 0) {
    float m = r1[0] / (float)(BB * LL);
    float v = r2[0] / (float)(BB * LL) - m * m;
    mean[d] = m; inv[d] = rsqrtf(v + 1e-5f);
  }
}

// x[b,l,c] = c<32 ? relu(bn(conv)) : pos
__global__ void k_build_x(const float* convpre, const float* mean, const float* inv,
                          const float* g, const float* be, const float* pos, float* x) {
  int idx = blockIdx.x * 256 + threadIdx.x;
  if (idx >= BB * LL * EE) return;
  int c = idx & 63, l = (idx >> 6) & 511, b = idx >> 15;
  float val;
  if (c < 32) {
    float v = convpre[((long)b * DD + c) * LL + l];
    val = fmaxf((v - mean[c]) * inv[c] * g[c] + be[c], 0.f);
  } else {
    val = pos[((long)b * LL + l) * DD + (c - 32)];
  }
  x[idx] = val;
}

// softmax over rows of (B*H*L, 512), scaled, write f16
__global__ void k_softmax(const float* att, _Float16* attH) {
  int row = blockIdx.x, lane = threadIdx.x;
  const float* a = att + (long)row * LL;
  _Float16* o = attH + (long)row * LL;
  const float scale = 0.17677669529663687f;   // 1/sqrt(32)
  float vals[16], mx = -1e30f;
#pragma unroll
  for (int i = 0; i < 16; ++i) { vals[i] = a[lane + i * 32] * scale; mx = fmaxf(mx, vals[i]); }
  for (int s = 16; s > 0; s >>= 1) mx = fmaxf(mx, __shfl_xor(mx, s, 32));
  float sum = 0.f;
#pragma unroll
  for (int i = 0; i < 16; ++i) { vals[i] = __expf(vals[i] - mx); sum += vals[i]; }
  for (int s = 16; s > 0; s >>= 1) sum += __shfl_xor(sum, s, 32);
  float r = 1.f / sum;
#pragma unroll
  for (int i = 0; i < 16; ++i) o[lane + i * 32] = (_Float16)(vals[i] * r);
}

// x = LN(x + y); also write f16 copy. 64 threads per token.
__global__ void k_add_ln(float* x, const float* y, const float* g, const float* be,
                         _Float16* xH) {
  int t = blockIdx.x, j = threadIdx.x;
  long off = (long)t * EE + j;
  float v = x[off] + y[off];
  __shared__ float red[64];
  red[j] = v; __syncthreads();
  for (int o = 32; o > 0; o >>= 1) { if (j < o) red[j] += red[j + o]; __syncthreads(); }
  float m = red[0] / 64.f; __syncthreads();
  float d = v - m;
  red[j] = d * d; __syncthreads();
  for (int o = 32; o > 0; o >>= 1) { if (j < o) red[j] += red[j + o]; __syncthreads(); }
  float inv = rsqrtf(red[0] / 64.f + 1e-5f);
  float out = d * inv * g[j] + be[j];
  x[off] = out; xH[off] = (_Float16)out;
}

// factored pairwise projections: u = x@W_A + pos@W_C + b/2 ; v = x@W_B + pos@W_D + b/2
__global__ void k_uv(const float* x, const float* pos, const float* c1w, const float* c1b,
                     float* u, float* v) {
  int t = blockIdx.x, o = threadIdx.x;
  const float* xr = x + (long)t * EE;
  const float* pr = pos + (long)t * DD;
  float su = 0.5f * c1b[o], sv = su;
  for (int c = 0; c < 64; ++c) { su += xr[c] * c1w[c * 32 + o]; sv += xr[c] * c1w[(64 + c) * 32 + o]; }
  for (int c = 0; c < 32; ++c) { su += pr[c] * c1w[(128 + c) * 32 + o]; sv += pr[c] * c1w[(160 + c) * 32 + o]; }
  u[(long)t * 32 + o] = su; v[(long)t * 32 + o] = sv;
}

// BN stats of h1_pre = u[min]+v[max] over (b,i,j) per channel (32), via atomics
__global__ void k_pair_stats1(const float* u, const float* v, float* acc) {
  int b = blockIdx.y;
  int idx = blockIdx.x * 256 + threadIdx.x;
  int i = idx >> 9, j = idx & 511;
  int mn = i < j ? i : j, mx = i < j ? j : i;
  const float* up = u + ((long)b * LL + mn) * 32;
  const float* vp = v + ((long)b * LL + mx) * 32;
  __shared__ float r1[256], r2[256];
  for (int o = 0; o < 32; ++o) {
    float h = up[o] + vp[o];
    r1[threadIdx.x] = h; r2[threadIdx.x] = h * h; __syncthreads();
    for (int s = 128; s > 0; s >>= 1) {
      if (threadIdx.x < s) { r1[threadIdx.x] += r1[threadIdx.x + s]; r2[threadIdx.x] += r2[threadIdx.x + s]; }
      __syncthreads();
    }
    if (threadIdx.x == 0) { atomicAdd(&acc[o], r1[0]); atomicAdd(&acc[32 + o], r2[0]); }
    __syncthreads();
  }
}

__global__ void k_finalize(const float* acc, float* mean, float* inv, float cnt) {
  int o = threadIdx.x;
  float m = acc[o] / cnt;
  float var = acc[32 + o] / cnt - m * m;
  mean[o] = m; inv[o] = rsqrtf(var + 1e-5f);
}

// BN stats of h2_pre = bn1(h1)relu @ c2_w + c2_b
__global__ void k_pair_stats2(const float* u, const float* v,
                              const float* mean1, const float* inv1,
                              const float* g1, const float* b1,
                              const float* c2w, const float* c2b, float* acc) {
  __shared__ float ws[1024];
  for (int i = threadIdx.x; i < 1024; i += 256) ws[i] = c2w[i];
  __syncthreads();
  int b = blockIdx.y;
  int idx = blockIdx.x * 256 + threadIdx.x;
  int i = idx >> 9, j = idx & 511;
  int mn = i < j ? i : j, mx = i < j ? j : i;
  const float* up = u + ((long)b * LL + mn) * 32;
  const float* vp = v + ((long)b * LL + mx) * 32;
  float h1[32];
  for (int o = 0; o < 32; ++o)
    h1[o] = fmaxf(((up[o] + vp[o]) - mean1[o]) * inv1[o] * g1[o] + b1[o], 0.f);
  __shared__ float r1[256], r2[256];
  for (int p = 0; p < 32; ++p) {
    float s = c2b[p];
    for (int o = 0; o < 32; ++o) s += h1[o] * ws[o * 32 + p];
    r1[threadIdx.x] = s; r2[threadIdx.x] = s * s; __syncthreads();
    for (int st = 128; st > 0; st >>= 1) {
      if (threadIdx.x < st) { r1[threadIdx.x] += r1[threadIdx.x + st]; r2[threadIdx.x] += r2[threadIdx.x + st]; }
      __syncthreads();
    }
    if (threadIdx.x == 0) { atomicAdd(&acc[p], r1[0]); atomicAdd(&acc[32 + p], r2[0]); }
    __syncthreads();
  }
}

// final score + S + P_hat + constraint matrix m
__global__ void k_score(const float* u, const float* v,
                        const float* mean1, const float* inv1, const float* g1, const float* b1,
                        const float* c2w, const float* c2b,
                        const float* mean2, const float* inv2, const float* g2, const float* b2,
                        const float* c3w, const float* c3b, const float* s_ptr,
                        const float* seq, float* score_out, float* S, float* Ph, float* mmat) {
  __shared__ float ws[1024];
  for (int i = threadIdx.x; i < 1024; i += 256) ws[i] = c2w[i];
  __syncthreads();
  int b = blockIdx.y;
  int idx = blockIdx.x * 256 + threadIdx.x;
  int i = idx >> 9, j = idx & 511;
  int mn = i < j ? i : j, mx = i < j ? j : i;
  const float* up = u + ((long)b * LL + mn) * 32;
  const float* vp = v + ((long)b * LL + mx) * 32;
  float h1[32];
  for (int o = 0; o < 32; ++o)
    h1[o] = fmaxf(((up[o] + vp[o]) - mean1[o]) * inv1[o] * g1[o] + b1[o], 0.f);
  float sc = c3b[0];
  for (int p = 0; p < 32; ++p) {
    float s2 = c2b[p];
    for (int o = 0; o < 32; ++o) s2 += h1[o] * ws[o * 32 + p];
    float h2v = fmaxf((s2 - mean2[p]) * inv2[p] * g2[p] + b2[p], 0.f);
    sc += h2v * c3w[p];
  }
  long gidx = (long)b * LSQ + idx;
  const float* si = seq + ((long)b * LL + i) * 4;
  const float* sj = seq + ((long)b * LL + j) * 4;
  float mval = si[0] * sj[1] + sj[0] * si[1]   // A-U
             + si[2] * sj[3] + sj[2] * si[3]   // C-G
             + si[1] * sj[3] + sj[1] * si[3];  // U-G
  int dd = i - j; if (dd < 0) dd = -dd;
  if (dd <= 3) mval = 0.f;
  float sconst = s_ptr[0];
  float Sv = (1.f / (1.f + __expf(-(sc - sconst)))) * sc;
  float phv = (1.f / (1.f + __expf(-Sv))) * (1.f / (1.f + __expf(-(Sv - sconst))));
  score_out[gidx] = sc; S[gidx] = Sv; Ph[gidx] = phv; mmat[gidx] = mval;
}

__global__ void k_rowsum(const float* P, float* r) {
  int row = blockIdx.x;
  const float* p = P + (long)row * LL;
  float s = 0.f;
  for (int i = threadIdx.x; i < LL; i += 32) s += p[i];
  for (int o = 16; o > 0; o >>= 1) s += __shfl_xor(s, o, 32);
  if (threadIdx.x == 0) r[row] = s;
}

__global__ void k_minit(const float* r, const float* w_ptr, float* M) {
  int i = blockIdx.x * 256 + threadIdx.x;
  if (i < BB * LL) M[i] = w_ptr[0] * fmaxf(r[i] - 1.f, 0.f);
}

// Ph <- clip(relu(|Ph - lr*grad| - rho*lr), 0, 1)
__global__ void k_ph(float* Ph, const float* S, const float* m, const float* r,
                     const float* M, const float* rho, const float* alpha,
                     const float* da, int t) {
  long idx = (long)blockIdx.x * 256 + threadIdx.x;
  if (idx >= BLL) return;
  int j = idx & 511, i = (idx >> 9) & 511, b = (int)(idx >> 18);
  float lr = alpha[0] * powf(da[0], (float)t);
  float ri = r[b * LL + i], rj = r[b * LL + j];
  float sig_i = 1.f / (1.f + __expf(-(ri - 1.f)));
  float sig_j = 1.f / (1.f + __expf(-(rj - 1.f)));
  float gsum = -S[idx] + M[b * LL + i] * sig_i + M[b * LL + j] * sig_j;  // gP + gP^T (S symmetric)
  float ph = Ph[idx];
  float grad = ph * m[idx] * gsum;
  float val = fabsf(ph - lr * grad) - rho[i * LL + j] * lr;
  Ph[idx] = fminf(fmaxf(val, 0.f), 1.f);
}

// P = 0.5*(Ph^2 + (Ph^T)^2) * m
__global__ void k_contact(const float* Ph, const float* m, float* P) {
  long idx = (long)blockIdx.x * 256 + threadIdx.x;
  if (idx >= BLL) return;
  int j = idx & 511, i = (idx >> 9) & 511, b = (int)(idx >> 18);
  float p1 = Ph[idx];
  float p2 = Ph[(long)b * LSQ + (long)j * LL + i];
  P[idx] = 0.5f * (p1 * p1 + p2 * p2) * m[idx];
}

__global__ void k_mupdate(const float* r, float* M, const float* beta,
                          const float* db, int t) {
  int i = blockIdx.x * 256 + threadIdx.x;
  if (i < BB * LL) M[i] += beta[0] * powf(db[0], (float)t) * fmaxf(r[i] - 1.f, 0.f);
}

// ---------------------------------------------------------------------------
extern "C" void kernel_launch(void* const* d_in, const int* in_sizes, int n_in,
                              void* d_out, int out_size, void* d_ws, size_t ws_size,
                              hipStream_t stream) {
  auto F = [&](int i) { return (const float*)d_in[i]; };
  // input indices (setup_inputs insertion order, params flattened in dict order)
  const float* pe    = F(0);
  const float* seq   = F(1);
  const float* convw = F(2);  const float* convb = F(3);
  const float* bn1g  = F(4);  const float* bn1b  = F(5);
  const float* pew1  = F(6);  const float* peb1  = F(7);
  const float* pew2  = F(8);  const float* peb2  = F(9);
  const float* pew3  = F(10); const float* peb3  = F(11);
  // layers: 12 + 16*li + {wq,wk,wv,wo,bq,bk,bv,bo,w1,b1,w2,b2,g1,be1,g2,be2}
  const float* c1w = F(60); const float* c1b = F(61);
  const float* g1c = F(62); const float* b1c = F(63);
  const float* c2w = F(64); const float* c2b = F(65);
  const float* g2c = F(66); const float* b2c = F(67);
  const float* c3w = F(68); const float* c3b = F(69);
  const float* s_p = F(70); const float* w_p = F(71);
  const float* rho = F(72);
  const float* alpha = F(73); const float* beta = F(74);
  const float* da = F(75);    const float* db = F(76);

  // workspace bump allocator
  char* wp = (char*)d_ws;
  auto alloc = [&](size_t bytes) { void* p = wp; wp += (bytes + 255) & ~(size_t)255; return p; };
  float*     pos     = (float*)alloc(BB*LL*DD*4);
  float*     convpre = (float*)alloc(BB*DD*LL*4);
  float*     x       = (float*)alloc(BB*LL*EE*4);
  _Float16*  xH      = (_Float16*)alloc(BB*LL*EE*2);
  float*     y       = (float*)alloc(BB*LL*EE*4);
  _Float16*  qH      = (_Float16*)alloc(BB*LL*EE*2);
  _Float16*  kH      = (_Float16*)alloc(BB*LL*EE*2);
  _Float16*  vH      = (_Float16*)alloc(BB*LL*EE*2);
  _Float16*  oH      = (_Float16*)alloc(BB*LL*EE*2);
  float*     att     = (float*)alloc((size_t)BB*HH*LSQ*4);
  _Float16*  attH    = (_Float16*)alloc((size_t)BB*HH*LSQ*2);
  _Float16*  ff1H    = (_Float16*)alloc((size_t)BB*LL*FFD*2);
  _Float16*  wHpool  = (_Float16*)alloc((size_t)3*(4*4096 + 2*131072)*2);
  float*     u       = (float*)alloc(BB*LL*DD*4);
  float*     v       = (float*)alloc(BB*LL*DD*4);
  float*     bn1m    = (float*)alloc(32*4);
  float*     bn1i    = (float*)alloc(32*4);
  float*     acc1    = (float*)alloc(64*4);
  float*     mean1   = (float*)alloc(32*4);
  float*     inv1    = (float*)alloc(32*4);
  float*     acc2    = (float*)alloc(64*4);
  float*     mean2   = (float*)alloc(32*4);
  float*     inv2    = (float*)alloc(32*4);
  float*     S       = (float*)alloc((size_t)BLL*4);
  float*     mmat    = (float*)alloc((size_t)BLL*4);
  float*     Ph      = (float*)alloc((size_t)BLL*4);
  float*     P       = (float*)alloc((size_t)BLL*4);
  float*     r       = (float*)alloc(BB*LL*4);
  float*     Mbuf    = (float*)alloc(BB*LL*4);

  float* score_out = (float*)d_out;            // first BLL floats
  float* Pf_out    = (float*)d_out + BLL;      // second BLL floats

  // ---- convert transformer weights to f16 once per call ----
  _Float16* wH[3][6];
  {
    _Float16* cur = wHpool;
    for (int li = 0; li < 3; ++li) {
      int base = 12 + 16 * li;
      int idxs[6] = { base + 0, base + 1, base + 2, base + 3, base + 8, base + 10 };
      int szs[6]  = { 4096, 4096, 4096, 4096, 131072, 131072 };
      for (int t2 = 0; t2 < 6; ++t2) {
        wH[li][t2] = cur;
        k_cvt_f16<<<(szs[t2] + 255) / 256, 256, 0, stream>>>(cur, F(idxs[t2]), szs[t2]);
        cur += szs[t2];
      }
    }
  }

  // ---- front end ----
  k_pe_mlp<<<BB * LL, 160, 0, stream>>>(pe, pew1, peb1, pew2, peb2, pew3, peb3, pos);
  k_conv<<<(BB * DD * LL + 255) / 256, 256, 0, stream>>>(seq, convw, convb, convpre);
  k_bn1d_stats<<<DD, 256, 0, stream>>>(convpre, bn1m, bn1i);
  k_build_x<<<(BB * LL * EE + 255) / 256, 256, 0, stream>>>(convpre, bn1m, bn1i, bn1g, bn1b, pos, x);
  k_cvt_f16<<<(BB * LL * EE + 255) / 256, 256, 0, stream>>>(xH, x, BB * LL * EE);

  // ---- 3 transformer layers ----
  const long sTok = (long)LL * EE;   // per-batch token-matrix stride
  for (int li = 0; li < 3; ++li) {
    int base = 12 + 16 * li;
    const float *bq = F(base + 4), *bk = F(base + 5), *bv = F(base + 6), *bo = F(base + 7);
    const float *b1 = F(base + 9), *b2 = F(base + 11);
    const float *g1 = F(base + 12), *be1 = F(base + 13), *g2 = F(base + 14), *be2 = F(base + 15);
    // QKV projections (M=1024, N=64, K=64) -> f16
    k_gemm<false><<<dim3(16, 2, 1), 256, 0, stream>>>(xH, 64, 0, 0, wH[li][0], 64, 0, 0,
        nullptr, qH, 64, 0, 0, bq, 1, 1024, 64, 64, 0);
    k_gemm<false><<<dim3(16, 2, 1), 256, 0, stream>>>(xH, 64, 0, 0, wH[li][1], 64, 0, 0,
        nullptr, kH, 64, 0, 0, bk, 1, 1024, 64, 64, 0);
    k_gemm<false><<<dim3(16, 2, 1), 256, 0, stream>>>(xH, 64, 0, 0, wH[li][2], 64, 0, 0,
        nullptr, vH, 64, 0, 0, bv, 1, 1024, 64, 64, 0);
    // scores = q @ k^T per (b,h): M=N=512, K=32, batch=4
    k_gemm<true><<<dim3(8, 16, 4), 256, 0, stream>>>(qH, 64, sTok, DHD, kH, 64, sTok, DHD,
        att, nullptr, 512, 2L * LSQ, (long)LSQ, nullptr, HH, 512, 512, 32, 0);
    k_softmax<<<BB * HH * LL, 32, 0, stream>>>(att, attH);
    // o = att @ v : M=512, N=32, K=512, batch=4
    k_gemm<false><<<dim3(8, 1, 4), 256, 0, stream>>>(attH, 512, 2L * LSQ, (long)LSQ,
        vH, 64, sTok, DHD, nullptr, oH, 64, sTok, DHD, nullptr, HH, 512, 32, 512, 0);
    // out proj: y = o @ wo + bo
    k_gemm<false><<<dim3(16, 2, 1), 256, 0, stream>>>(oH, 64, 0, 0, wH[li][3], 64, 0, 0,
        y, nullptr, 64, 0, 0, bo, 1, 1024, 64, 64, 0);
    k_add_ln<<<BB * LL, 64, 0, stream>>>(x, y, g1, be1, xH);
    // FF1: relu(x @ w1 + b1) -> f16 (M=1024, N=2048, K=64)
    k_gemm<false><<<dim3(16, 64, 1), 256, 0, stream>>>(xH, 64, 0, 0, wH[li][4], 2048, 0, 0,
        nullptr, ff1H, 2048, 0, 0, b1, 1, 1024, 2048, 64, 1);
    // FF2: y = ff1 @ w2 + b2 (M=1024, N=64, K=2048)
    k_gemm<false><<<dim3(16, 2, 1), 256, 0, stream>>>(ff1H, 2048, 0, 0, wH[li][5], 64, 0, 0,
        y, nullptr, 64, 0, 0, b2, 1, 1024, 64, 2048, 0);
    k_add_ln<<<BB * LL, 64, 0, stream>>>(x, y, g2, be2, xH);
  }

  // ---- factored pairwise head ----
  k_uv<<<BB * LL, 32, 0, stream>>>(x, pos, c1w, c1b, u, v);
  k_zero<<<1, 64, 0, stream>>>(acc1, 64);
  k_pair_stats1<<<dim3(LSQ / 256, BB, 1), 256, 0, stream>>>(u, v, acc1);
  k_finalize<<<1, 32, 0, stream>>>(acc1, mean1, inv1, (float)BLL);
  k_zero<<<1, 64, 0, stream>>>(acc2, 64);
  k_pair_stats2<<<dim3(LSQ / 256, BB, 1), 256, 0, stream>>>(u, v, mean1, inv1, g1c, b1c,
                                                            c2w, c2b, acc2);
  k_finalize<<<1, 32, 0, stream>>>(acc2, mean2, inv2, (float)BLL);
  k_score<<<dim3(LSQ / 256, BB, 1), 256, 0, stream>>>(u, v, mean1, inv1, g1c, b1c, c2w, c2b,
      mean2, inv2, g2c, b2c, c3w, c3b, s_p, seq, score_out, S, Ph, mmat);

  // ---- iterative post-processing ----
  int gEl = (BLL + 255) / 256;
  k_contact<<<gEl, 256, 0, stream>>>(Ph, mmat, P);
  k_rowsum<<<BB * LL, 32, 0, stream>>>(P, r);
  k_minit<<<(BB * LL + 255) / 256, 256, 0, stream>>>(r, w_p, Mbuf);
  for (int t = 0; t < NSTEPS; ++t) {
    k_rowsum<<<BB * LL, 32, 0, stream>>>(P, r);
    k_ph<<<gEl, 256, 0, stream>>>(Ph, S, mmat, r, Mbuf, rho, alpha, da, t);
    k_contact<<<gEl, 256, 0, stream>>>(Ph, mmat, P);
    k_rowsum<<<BB * LL, 32, 0, stream>>>(P, r);
    k_mupdate<<<(BB * LL + 255) / 256, 256, 0, stream>>>(r, Mbuf, beta, db, t);
  }
  k_copy<<<gEl, 256, 0, stream>>>(Pf_out, P, BLL);
}
